// PaiAutoencoder_51402168599241
// MI455X (gfx1250) — matrix operand order repaired and checked by reading, not compiled, not run on hardware
//
#include <hip/hip_runtime.h>
#include <hip/hip_bf16.h>

// ---------------------------------------------------------------------------
// CDNA5 (gfx1250) WMMA types
// ---------------------------------------------------------------------------
typedef __attribute__((ext_vector_type(16))) __bf16          v16bf;
typedef __attribute__((ext_vector_type(16))) unsigned short  v16us;
typedef __attribute__((ext_vector_type(8)))  float           v8f;
typedef __attribute__((ext_vector_type(4)))  unsigned int    v4u;

__device__ __forceinline__ unsigned short f2bf(float f) {
    unsigned u = __float_as_uint(f);
    unsigned r = u + 0x7FFFu + ((u >> 16) & 1u);   // round-to-nearest-even
    return (unsigned short)(r >> 16);
}

// ---------------------------------------------------------------------------
// f32 -> bf16 converters (contiguous / padded-stride / padded+zp variants)
// ---------------------------------------------------------------------------
__global__ void k_f32_to_bf16(const float* __restrict__ src,
                              unsigned short* __restrict__ dst, long n) {
    long i = (long)blockIdx.x * blockDim.x + threadIdx.x;
    if (i < n) dst[i] = f2bf(src[i]);
}

// dst has row stride ldp >= cols; pad region zero-filled
__global__ void k_cvt_pad(const float* __restrict__ src,
                          unsigned short* __restrict__ dst,
                          long rows, int cols, int ldp) {
    long i = (long)blockIdx.x * blockDim.x + threadIdx.x;
    long tot = rows * ldp;
    if (i >= tot) return;
    long r = i / ldp;
    int  c = (int)(i - r * ldp);
    dst[i] = (c < cols) ? f2bf(src[r * cols + c]) : (unsigned short)0;
}

// activations (rows = B*Nv) with zp: zero vertex row Nv-1; padded stride
__global__ void k_cvt_zp_pad(const float* __restrict__ src,
                             unsigned short* __restrict__ dst,
                             long rows, int cols, int ldp, int Nv) {
    long i = (long)blockIdx.x * blockDim.x + threadIdx.x;
    long tot = rows * ldp;
    if (i >= tot) return;
    long r = i / ldp;
    int  c = (int)(i - r * ldp);
    float f = 0.f;
    if (c < cols && (r % Nv) != Nv - 1) f = src[r * cols + c];
    dst[i] = f2bf(f);
}

// ---------------------------------------------------------------------------
// Per-vertex basis softmax -> 9x9 adjacency weights
// ---------------------------------------------------------------------------
__global__ void k_adjw(const float* __restrict__ tv,      // N x 128
                       const float* __restrict__ tmpt_w,  // 128
                       const float* __restrict__ tmpt_b,  // 1
                       const float* __restrict__ mlp_w,   // 32 x 128
                       const float* __restrict__ mlp_b,   // 32
                       const float* __restrict__ adjw,    // 32 x 81
                       float* __restrict__ adj_out,       // N x 81
                       int Nv) {
    int n = blockIdx.x * blockDim.x + threadIdx.x;
    if (n >= Nv) return;
    const float* t = tv + (long)n * 128;
    float td = tmpt_b[0];
    for (int j = 0; j < 128; ++j) td += t[j] * tmpt_w[j];
    float tmpt = 1.f / (1.f + __expf(-td)) * 0.09f + 0.01f;
    float logits[32];
    float mx = -3.0e38f;
    for (int s = 0; s < 32; ++s) {
        float acc = mlp_b[s];
        const float* w = mlp_w + (long)s * 128;
        for (int j = 0; j < 128; ++j) acc += t[j] * w[j];
        acc /= tmpt;
        logits[s] = acc;
        mx = fmaxf(mx, acc);
    }
    float denom = 0.f;
    for (int s = 0; s < 32; ++s) { logits[s] = __expf(logits[s] - mx); denom += logits[s]; }
    float inv = 1.f / denom;
    float* out = adj_out + (long)n * 81;
    for (int e = 0; e < 81; ++e) {
        float acc = 0.f;
        for (int s = 0; s < 32; ++s) acc += logits[s] * adjw[(long)s * 81 + e];
        out[e] = acc * inv;
    }
}

// ---------------------------------------------------------------------------
// Neighbor gather + adjacency mix + ELU -> bf16 H, padded row stride ldh
// ---------------------------------------------------------------------------
__global__ void k_mix(const float* __restrict__ x,   // B x Nv x F
                      const int* __restrict__ nb,    // Nv x 9
                      const float* __restrict__ adj, // Nv x 81
                      unsigned short* __restrict__ H,
                      int Nv, int F, int bpb, int ldh) {
    __shared__ float sAdj[81];
    __shared__ int   sNb[9];
    int bid = blockIdx.x;
    int bn  = bid / bpb;
    int sub = bid - bn * bpb;
    int b = bn / Nv, n = bn - b * Nv;
    if (threadIdx.x < 81) sAdj[threadIdx.x] = adj[(long)n * 81 + threadIdx.x];
    if (threadIdx.x < 9)  sNb[threadIdx.x]  = nb[(long)n * 9 + threadIdx.x];
    __syncthreads();
    int ftot = F * 9;
    int ft = sub * 128 + threadIdx.x;
    if (ft >= ftot) return;
    int f = ft / 9, t = ft - f * 9;
    const float* xb = x + (long)b * Nv * F;
    float acc = 0.f;
#pragma unroll
    for (int k = 0; k < 9; ++k) {
        int j = sNb[k];
        float xv = (j == Nv - 1) ? 0.f : xb[(long)j * F + f];
        acc += xv * sAdj[k * 9 + t];
    }
    acc = acc > 0.f ? acc : expm1f(acc);
    H[(long)bn * ldh + ft] = f2bf(acc);
}

// ---------------------------------------------------------------------------
// bf16 WMMA GEMM: 64x64 block tile, 4 waves x (2x2) 16x16 WMMA tiles,
// double-buffered fragment-ordered LDS, vectorized b128 staging.
//   C[M,N] = A[M,K] * B      (B is [K,N] row-major, or [N,K] if TRANSB)
// Operand row strides (lda/ldb) are expected padded to multiples of 8.
// ---------------------------------------------------------------------------
enum { GF_TRANSB = 1, GF_ELU = 2, GF_BIAS = 4, GF_SKIP = 8, GF_ZROW = 16 };

__global__ __launch_bounds__(128)
void k_gemm_bf16_wmma(const unsigned short* __restrict__ A,
                      const unsigned short* __restrict__ Bm,
                      const float* __restrict__ bias,
                      const float* __restrict__ skip,
                      float* __restrict__ C,
                      int M, int N, int K,
                      int lda, int ldb, int ldc,
                      long strideA, long strideB, long strideC,
                      int flags, int zperiod) {
    __shared__ v16us sA[2][4][32];   // [buf][rowhalf][lane], fragment order
    __shared__ v16us sB[2][4][32];   // [buf][colhalf][lane], fragment order

    const int tid = threadIdx.x;
    const int m0 = blockIdx.y * 64, n0 = blockIdx.x * 64;
    const unsigned short* Ab = A  + (long)blockIdx.z * strideA;
    const unsigned short* Bb = Bm + (long)blockIdx.z * strideB;
    float* Cb = C + (long)blockIdx.z * strideC;
    const float* skipb = skip ? (skip + (long)blockIdx.z * strideC) : skip;

    const int w    = tid >> 5;
    const int lane = tid & 31;
    const int grp  = lane >> 4;
    const int ml   = lane & 15;
    const int wr   = (w >> 1) * 2;
    const int wc   = (w & 1) * 2;

    const bool transB = (flags & GF_TRANSB) != 0;
    const bool aVec = ((lda & 7) == 0);
    const bool bVec = ((ldb & 7) == 0);

    // stage one 64x32 A tile + 32x64 B tile (k-offset k0) into buffer `buf`
    auto stage = [&](int k0, int buf) {
        unsigned short* dA = (unsigned short*)&sA[buf][0][0];
        unsigned short* dB = (unsigned short*)&sB[buf][0][0];
        // ---- A: 8-half chunks land contiguously in one fragment slot ----
        if (aVec) {
            for (int q = tid; q < 256; q += 128) {
                int r = q >> 2, c8 = (q & 3) * 8;
                int gr = m0 + r, gc = k0 + c8;
                int g  = c8 >> 4;
                int i0 = c8 & 8;
                int slot = (((r >> 4) * 32) + (r & 15) + 16 * g) * 16 + i0;
                if (gr < M && gc + 8 <= K) {
                    *(v4u*)&dA[slot] = *(const v4u*)&Ab[(long)gr * lda + gc];
                } else {
#pragma unroll
                    for (int j = 0; j < 8; ++j)
                        dA[slot + j] = (gr < M && gc + j < K)
                                           ? Ab[(long)gr * lda + gc + j]
                                           : (unsigned short)0;
                }
            }
        } else {
            for (int e = tid; e < 2048; e += 128) {
                int r = e >> 5, c = e & 31;
                int gr = m0 + r, gc = k0 + c;
                unsigned short v = (gr < M && gc < K)
                                       ? Ab[(long)gr * lda + gc] : (unsigned short)0;
                int g = c >> 4;
                int i = c & 15;
                dA[(((r >> 4) * 32) + (r & 15) + 16 * g) * 16 + i] = v;
            }
        }
        // ---- B ----
        if (transB) {                       // B is [N x K]: contiguous in k
            if (bVec) {
                for (int q = tid; q < 256; q += 128) {
                    int n = q >> 2, c8 = (q & 3) * 8;
                    int gn = n0 + n, gk = k0 + c8;
                    int slot = (((n >> 4) * 32) + (n & 15) + 16 * (c8 >> 4)) * 16
                               + (c8 & 8);
                    if (gn < N && gk + 8 <= K) {
                        *(v4u*)&dB[slot] = *(const v4u*)&Bb[(long)gn * ldb + gk];
                    } else {
#pragma unroll
                        for (int j = 0; j < 8; ++j)
                            dB[slot + j] = (gn < N && gk + j < K)
                                               ? Bb[(long)gn * ldb + gk + j]
                                               : (unsigned short)0;
                    }
                }
            } else {
                for (int e = tid; e < 2048; e += 128) {
                    int k = e >> 6, c = e & 63;
                    int gn = n0 + c, gk = k0 + k;
                    unsigned short v = (gn < N && gk < K)
                                           ? Bb[(long)gn * ldb + gk] : (unsigned short)0;
                    dB[(((c >> 4) * 32) + (c & 15) + 16 * (k >> 4)) * 16 + (k & 15)] = v;
                }
            }
        } else {                            // B is [K x N]: contiguous in n
            if (bVec) {
                for (int q = tid; q < 256; q += 128) {
                    int k = q >> 3, c8 = (q & 7) * 8;
                    int gk = k0 + k, gn0 = n0 + c8;
                    if (gk < K && gn0 + 8 <= N) {
                        v4u val = *(const v4u*)&Bb[(long)gk * ldb + gn0];
#pragma unroll
                        for (int d = 0; d < 4; ++d) {
                            unsigned wv = val[d];
                            int n1 = c8 + 2 * d;
                            int s1 = (((n1 >> 4) * 32) + (n1 & 15) + 16 * (k >> 4)) * 16
                                     + (k & 15);
                            int n2 = n1 + 1;
                            int s2 = (((n2 >> 4) * 32) + (n2 & 15) + 16 * (k >> 4)) * 16
                                     + (k & 15);
                            dB[s1] = (unsigned short)(wv & 0xffffu);
                            dB[s2] = (unsigned short)(wv >> 16);
                        }
                    } else {
#pragma unroll
                        for (int j = 0; j < 8; ++j) {
                            int n = c8 + j;
                            unsigned short v = (gk < K && n0 + n < N)
                                                   ? Bb[(long)gk * ldb + n0 + n]
                                                   : (unsigned short)0;
                            dB[(((n >> 4) * 32) + (n & 15) + 16 * (k >> 4)) * 16
                               + (k & 15)] = v;
                        }
                    }
                }
            } else {
                for (int e = tid; e < 2048; e += 128) {
                    int k = e >> 6, c = e & 63;
                    int gk = k0 + k, gn = n0 + c;
                    unsigned short v = (gk < K && gn < N)
                                           ? Bb[(long)gk * ldb + gn] : (unsigned short)0;
                    dB[(((c >> 4) * 32) + (c & 15) + 16 * (k >> 4)) * 16 + (k & 15)] = v;
                }
            }
        }
    };

    v8f acc00 = {}, acc01 = {}, acc10 = {}, acc11 = {};

    stage(0, 0);
    __syncthreads();
    int pb = 0;
    for (int k0 = 0; k0 < K; k0 += 32) {
        // fragment vector loads from current buffer (ds_load_b128 pairs)
        v16us a0u = sA[pb][wr + 0][lane];
        v16us a1u = sA[pb][wr + 1][lane];
        v16us b0u = sB[pb][wc + 0][lane];
        v16us b1u = sB[pb][wc + 1][lane];

        // speculative prefetch two tiles ahead
        if (k0 + 64 < K) {
            int pr = m0 + (tid >> 1);
            if (pr < M && (tid & 1) == 0)
                __builtin_prefetch(&Ab[(long)pr * lda + (k0 + 64)], 0, 1);
        }
        // software pipeline: stage next tile into the other buffer
        if (k0 + 32 < K) stage(k0 + 32, pb ^ 1);

        v16bf a0 = __builtin_bit_cast(v16bf, a0u);
        v16bf a1 = __builtin_bit_cast(v16bf, a1u);
        v16bf b0 = __builtin_bit_cast(v16bf, b0u);
        v16bf b1 = __builtin_bit_cast(v16bf, b1u);
        acc00 = __builtin_amdgcn_wmma_f32_16x16x32_bf16(false, a0, false, b0,
                                                        (short)0, acc00, false, false);
        acc01 = __builtin_amdgcn_wmma_f32_16x16x32_bf16(false, a0, false, b1,
                                                        (short)0, acc01, false, false);
        acc10 = __builtin_amdgcn_wmma_f32_16x16x32_bf16(false, a1, false, b0,
                                                        (short)0, acc10, false, false);
        acc11 = __builtin_amdgcn_wmma_f32_16x16x32_bf16(false, a1, false, b1,
                                                        (short)0, acc11, false, false);
        __syncthreads();
        pb ^= 1;
    }

    // ---- epilogue ----
    const v8f* accs[2][2] = {{&acc00, &acc01}, {&acc10, &acc11}};
#pragma unroll
    for (int ti = 0; ti < 2; ++ti) {
#pragma unroll
        for (int tj = 0; tj < 2; ++tj) {
            int col = n0 + (wc + tj) * 16 + ml;
            if (col >= N) continue;
            float bv = (flags & GF_BIAS) ? bias[col] : 0.f;
            const v8f& a = *accs[ti][tj];
#pragma unroll
            for (int r = 0; r < 8; ++r) {
                int row = m0 + (wr + ti) * 16 + r + 8 * grp;
                if (row < M) {
                    float v = a[r] + bv;
                    if (flags & GF_ELU)  v = v > 0.f ? v : expm1f(v);
                    if ((flags & GF_ZROW) && ((row % zperiod) == zperiod - 1)) v = 0.f;
                    if (flags & GF_SKIP) v += skipb[(long)row * ldc + col];
                    Cb[(long)row * ldc + col] = v;
                }
            }
        }
    }
}

// ---------------------------------------------------------------------------
// Host orchestration
// ---------------------------------------------------------------------------
static inline int cdiv_i(int a, int b) { return (a + b - 1) / b; }
static inline int pad8(int x) { return (x + 7) & ~7; }

extern "C" void kernel_launch(void* const* d_in, const int* in_sizes, int n_in,
                              void* d_out, int out_size, void* d_ws, size_t ws_size,
                              hipStream_t stream) {
    (void)in_sizes; (void)n_in; (void)out_size; (void)ws_size;
    const int NSv[4] = {8193, 2049, 513, 129};
    const int Bb = 8;

    // ---- input unpack (setup_inputs dict order, lists/dicts flattened) ----
    const float* x_in = (const float*)d_in[0];
    auto LP = [&](int base, int j) -> const float* { return (const float*)d_in[base + j]; };
    // per-layer slots: 0 conv_w 1 conv_b 2 mlp_out_w 3 mlp_out_b
    //                  4 tmpt_w 5 tmpt_b 6 mlp_w 7 mlp_b 8 adjw
    const float* fc_enc_w = (const float*)d_in[64];
    const float* fc_enc_b = (const float*)d_in[65];
    const float* fc_dec_w = (const float*)d_in[66];
    const float* fc_dec_b = (const float*)d_in[67];
    const float* Dm[3] = {(const float*)d_in[68], (const float*)d_in[69], (const float*)d_in[70]};
    const float* Um[3] = {(const float*)d_in[71], (const float*)d_in[72], (const float*)d_in[73]};
    const float* TV[3] = {(const float*)d_in[74], (const float*)d_in[75], (const float*)d_in[76]};
    const int*   NBp[3] = {(const int*)d_in[77], (const int*)d_in[78], (const int*)d_in[79]};

    // ---- workspace arena ----
    char* wsb = (char*)d_ws;
    size_t off = 0;
    auto arena = [&](size_t bytes) -> void* {
        void* p = (void*)(wsb + off);
        off += (bytes + 255) & ~(size_t)255;
        return p;
    };
    const long maxAct = (long)Bb * 8193 * 64;                   // 4.19M f32
    float* actA = (float*)arena((size_t)maxAct * 4);
    float* actB = (float*)arena((size_t)maxAct * 4);
    unsigned short* actBF = (unsigned short*)arena((size_t)maxAct * 2);
    unsigned short* Hbf   = (unsigned short*)arena((size_t)Bb * 8193 * 576 * 2);
    float* adjb = (float*)arena((size_t)8193 * 81 * 4);
    unsigned short* wbf = (unsigned short*)arena((size_t)8193 * 2056 * 2); // padded D/U
    float* zf = (float*)arena((size_t)Bb * 256 * 4);
    unsigned short* zbf = (unsigned short*)arena((size_t)Bb * 256 * 2);

    // ---- launch helpers ----
    auto cvt = [&](const float* s, unsigned short* d, long n) {
        k_f32_to_bf16<<<dim3((unsigned)((n + 255) / 256)), 256, 0, stream>>>(s, d, n);
    };
    auto cvt_pad = [&](const float* s, unsigned short* d, long rows, int cols, int ldp) {
        long n = rows * ldp;
        k_cvt_pad<<<dim3((unsigned)((n + 255) / 256)), 256, 0, stream>>>(s, d, rows, cols, ldp);
    };
    auto cvt_zp_pad = [&](const float* s, unsigned short* d, long rows, int cols,
                          int ldp, int Nv) {
        long n = rows * ldp;
        k_cvt_zp_pad<<<dim3((unsigned)((n + 255) / 256)), 256, 0, stream>>>(s, d, rows, cols,
                                                                            ldp, Nv);
    };
    auto gemm = [&](const unsigned short* A, const unsigned short* Bm,
                    const float* bias, const float* skip, float* C,
                    int M, int N, int K, int lda, int ldb, int ldc,
                    long sA, long sB, long sC, int nbatch, int flags, int zperiod) {
        dim3 g((unsigned)cdiv_i(N, 64), (unsigned)cdiv_i(M, 64), (unsigned)nbatch);
        k_gemm_bf16_wmma<<<g, 128, 0, stream>>>(A, Bm, bias, skip, C,
                                                M, N, K, lda, ldb, ldc,
                                                sA, sB, sC, flags, zperiod);
    };

    // one PAI conv layer: xcur (B,Nv,Fi) f32 -> out (B,Nv,Fo) f32
    auto pai_conv = [&](const float* xcur, int level, int pbase,
                        int Fi, int Fo, bool identity_act, float* out) {
        int Nv = NSv[level];
        int ldh = pad8(Fi * 9);
        int ldp = pad8(Fi);
        // per-vertex adjacency weights
        k_adjw<<<dim3((unsigned)cdiv_i(Nv, 128)), 128, 0, stream>>>(
            TV[level], LP(pbase, 4), LP(pbase, 5), LP(pbase, 6), LP(pbase, 7),
            LP(pbase, 8), adjb, Nv);
        // gather + mix + ELU -> bf16 H (padded stride)
        int ftot = Fi * 9, bpb = cdiv_i(ftot, 128);
        k_mix<<<dim3((unsigned)(Bb * Nv * bpb)), 128, 0, stream>>>(
            xcur, NBp[level], adjb, Hbf, Nv, Fi, bpb, ldh);
        // skip path: out = x_zp @ mlp_out_w^T + mlp_out_b
        cvt_zp_pad(xcur, actBF, (long)Bb * Nv, Fi, ldp, Nv);
        cvt_pad(LP(pbase, 2), wbf, Fo, Fi, ldp);
        gemm(actBF, wbf, LP(pbase, 3), nullptr, out,
             Bb * Nv, Fo, Fi, ldp, ldp, Fo, 0, 0, 0, 1,
             GF_TRANSB | GF_BIAS, 1);
        // conv path fused: elu(H @ conv_w^T + conv_b) * zp + skip(out)
        cvt_pad(LP(pbase, 0), wbf, Fo, Fi * 9, ldh);
        int fl = GF_TRANSB | GF_BIAS | GF_ZROW | GF_SKIP | (identity_act ? 0 : GF_ELU);
        gemm(Hbf, wbf, LP(pbase, 1), out, out,
             Bb * Nv, Fo, Fi * 9, ldh, ldh, Fo, 0, 0, 0, 1, fl, Nv);
    };

    // pooling: out[b] = L(P x Mv) @ x[b](Mv x F)   (F always multiple of 64)
    auto pool = [&](const float* xcur, const float* L, int P, int Mv, int F, float* out) {
        int ldap = pad8(Mv);
        cvt(xcur, actBF, (long)Bb * Mv * F);
        cvt_pad(L, wbf, P, Mv, ldap);
        gemm(wbf, actBF, nullptr, nullptr, out,
             P, F, Mv, ldap, F, F,
             0, (long)Mv * F, (long)P * F, Bb, 0, 1);
    };

    // ------------------------------ encoder ------------------------------
    const float* cur = x_in;
    const int Fi_e[3] = {3, 64, 128}, Fo_e[3] = {64, 128, 256};
    for (int i = 0; i < 3; ++i) {
        float* o  = (cur == actA) ? actB : actA;
        pai_conv(cur, i, 1 + 9 * i, Fi_e[i], Fo_e[i], false, o);
        float* o2 = (o == actA) ? actB : actA;
        pool(o, Dm[i], NSv[i + 1], NSv[i], Fo_e[i], o2);
        cur = o2;
    }

    // ------------------------------ bottleneck FC ------------------------
    const long fin = (long)NSv[3] * 256;  // 33024 (multiple of 8)
    cvt(cur, actBF, (long)Bb * fin);
    cvt(fc_enc_w, wbf, 256L * fin);
    gemm(actBF, wbf, fc_enc_b, nullptr, zf,
         Bb, 256, (int)fin, (int)fin, (int)fin, 256, 0, 0, 0, 1,
         GF_TRANSB | GF_BIAS, 1);
    cvt(zf, zbf, (long)Bb * 256);
    cvt(fc_dec_w, wbf, fin * 256L);
    float* xdec = (cur == actA) ? actB : actA;
    gemm(zbf, wbf, fc_dec_b, nullptr, xdec,
         Bb, (int)fin, 256, 256, 256, (int)fin, 0, 0, 0, 1,
         GF_TRANSB | GF_BIAS, 1);
    cur = xdec;  // viewed as (B, 129, 256)

    // ------------------------------ decoder ------------------------------
    const int Fd_i[4] = {256, 128, 64, 32}, Fd_o[4] = {128, 64, 32, 3};
    for (int i = 0; i < 3; ++i) {
        int L = 2 - i;
        float* o = (cur == actA) ? actB : actA;
        pool(cur, Um[L], NSv[L], NSv[L + 1], Fd_i[i], o);  // upsample
        float* o2 = (o == actA) ? actB : actA;
        pai_conv(o, L, 28 + 9 * i, Fd_i[i], Fd_o[i], false, o2);
        cur = o2;
    }
    // final conv, identity activation, straight into d_out (B,8193,3)
    pai_conv(cur, 0, 28 + 27, 32, 3, true, (float*)d_out);
}
